// UNet_vanilla_48490180772345
// MI455X (gfx1250) — matrix operand order repaired and checked by reading, not compiled, not run on hardware
//
#include <hip/hip_runtime.h>

// ---------------------------------------------------------------------------
// Graph-transformer forward for MI455X (gfx1250, wave32, WMMA).
// Heavy GEMMs (edge E-proj / WOe / edge MLP, node projections / MLP) run on
// v_wmma_f32_16x16x32_f16 with f32 accumulation; everything else is
// bandwidth-bound elementwise / LN / softmax kernels touching each tensor once.
// ---------------------------------------------------------------------------

typedef __attribute__((ext_vector_type(16))) _Float16 v16h;
typedef __attribute__((ext_vector_type(4)))  _Float16 v4h;
typedef __attribute__((ext_vector_type(8)))  float    v8f;

#define BB   16
#define NN   128
#define DD   128
#define HH   8
#define DHH  16
#define FF   512
#define NLAYERS 4
#define BN   (BB*NN)        // 2048 node rows
#define BNN  (BB*NN*NN)     // 262144 edge rows
#define CHUNK  32768
#define NCHUNK (BNN/CHUNK)  // 8
#define SLOPE 0.01f
#define EPSLN 1e-5f

// ---------------------------------------------------------------------------
// WMMA GEMM:  C[M,N] = epi( A[M,K] @ Bw[K,N] + bias[N] (+ Resid[M,N]) )
// A: f32 or f16 (converted to f16 in LDS), Bw: f16, accum f32.
// Block = 256 thr = 8 waves (4x2) -> 64x32 tile, K-step 32.
// M%64==0, N%32==0, K%32==0 (true for all call sites).
// RES is compile-time so the epilogue has no branches; LDS row pitch is
// 36 halves (72B) so packed v4h (b64) LDS stores stay 8B-aligned.
// ---------------------------------------------------------------------------
template<typename TA, typename TC, int ACT, bool RES>
__global__ __launch_bounds__(256) void k_gemm(
    const TA* __restrict__ A, const _Float16* __restrict__ Bw,
    const float* __restrict__ bias, const float* __restrict__ Resid,
    TC* __restrict__ C, int M, int N, int K)
{
  __shared__ __align__(16) _Float16 As[64][36];   // [m][k]
  __shared__ __align__(16) _Float16 Bs[32][36];   // [n][k]  (transposed tile)
  const int tid  = threadIdx.x;
  const int wave = tid >> 5;
  const int lane = tid & 31;
  const int wm   = wave >> 1;       // 0..3
  const int wn   = wave & 1;        // 0..1
  const int lr   = lane & 15;
  const int hi   = lane >> 4;       // lane half-select
  const int m0   = blockIdx.x * 64;
  const int n0   = blockIdx.y * 32;

  v8f acc;
#pragma unroll
  for (int r = 0; r < 8; ++r) acc[r] = 0.0f;

  // B staging map: thread -> (nn = tid&31, kq = tid>>5), 4 consecutive k each
  const int bnn = tid & 31;
  const int bkq = tid >> 5;

  for (int k0 = 0; k0 < K; k0 += 32) {
    // ---- stage A 64x32 -> f16, vector loads (b128/b64) + packed b64 LDS stores
#pragma unroll
    for (int it = 0; it < 2; ++it) {
      const int e = (tid + it * 256) * 4;        // element index, step 4
      const int r = e >> 5, c = e & 31;
      v4h hv;
      if constexpr (sizeof(TA) == 4) {
        const float4 f = *reinterpret_cast<const float4*>(
            A + (size_t)(m0 + r) * K + (k0 + c));
        hv[0] = (_Float16)f.x; hv[1] = (_Float16)f.y;
        hv[2] = (_Float16)f.z; hv[3] = (_Float16)f.w;
      } else {
        hv = *reinterpret_cast<const v4h*>(A + (size_t)(m0 + r) * K + (k0 + c));
      }
      *reinterpret_cast<v4h*>(&As[r][c]) = hv;
    }
    // ---- stage B 32x32 transposed: 4 coalesced b16 loads -> one b64 LDS store
    {
      v4h bv;
#pragma unroll
      for (int q = 0; q < 4; ++q)
        bv[q] = Bw[(size_t)(k0 + bkq * 4 + q) * N + (n0 + bnn)];
      *reinterpret_cast<v4h*>(&Bs[bnn][bkq * 4]) = bv;
    }
    __syncthreads();

    // A-frag 16x32: lanes 0-15 row M=lr; halves 0..7 -> K=hi*8+h, 8..15 -> K=16+hi*8+h
    v16h af, bf;
#pragma unroll
    for (int h = 0; h < 8; ++h) {
      af[h]     = As[wm * 16 + lr][hi * 8 + h];
      af[h + 8] = As[wm * 16 + lr][16 + hi * 8 + h];
    }
    // B-frag 32x16: col N=lr; halves 0..7 -> K=hi*8+h, 8..15 -> K=16+hi*8+h
#pragma unroll
    for (int h = 0; h < 8; ++h) {
      bf[h]     = Bs[wn * 16 + lr][hi * 8 + h];
      bf[h + 8] = Bs[wn * 16 + lr][16 + hi * 8 + h];
    }

    acc = __builtin_amdgcn_wmma_f32_16x16x32_f16(false, af, false, bf,
                                                 (short)0, acc, false, false);
    __syncthreads();
  }

  // ---- epilogue: branch-free, incremental addressing ----------------------
  const int col = n0 + wn * 16 + lr;
  const float bv = bias[col];
  size_t off = (size_t)(m0 + wm * 16 + hi * 8) * (size_t)N + col;
#pragma unroll
  for (int r = 0; r < 8; ++r) {
    float v = acc[r] + bv;
    if constexpr (ACT == 1) v = (v > 0.0f) ? v : SLOPE * v;   // LeakyReLU
    if constexpr (RES) v += Resid[off];
    C[off] = (TC)v;
    off += (size_t)N;
  }
}

// ---------------------------------------------------------------------------
// Elementwise / LN / attention kernels (128 threads = one D-row)
// ---------------------------------------------------------------------------

// y = LN(x (+ pt[b])) * g + b ; one block per row, D=128
__global__ __launch_bounds__(128) void k_layernorm(
    const float* __restrict__ X, const float* __restrict__ pt,
    const float* __restrict__ g, const float* __restrict__ bta,
    float* __restrict__ Y, int rowsPerB)
{
  __shared__ float sh[128];
  const int row = blockIdx.x, t = threadIdx.x;
  const int b = row / rowsPerB;
  float v = X[(size_t)row * DD + t];
  if (pt) v += pt[b * DD + t];
  sh[t] = v; __syncthreads();
  for (int s = 64; s > 0; s >>= 1) { if (t < s) sh[t] += sh[t + s]; __syncthreads(); }
  const float mean = sh[0] * (1.0f / DD);
  __syncthreads();
  const float d = v - mean;
  sh[t] = d * d; __syncthreads();
  for (int s = 64; s > 0; s >>= 1) { if (t < s) sh[t] += sh[t + s]; __syncthreads(); }
  const float var = sh[0] * (1.0f / DD);
  Y[(size_t)row * DD + t] = d * rsqrtf(var + EPSLN) * g[t] + bta[t];
}

// node embed: x = x_t @ Wa + ba + pe[n]
__global__ __launch_bounds__(128) void k_node_embed(
    const float* __restrict__ xt, const float* __restrict__ Wa,
    const float* __restrict__ ba, const float* __restrict__ pe,
    float* __restrict__ X)
{
  const int row = blockIdx.x, t = threadIdx.x;
  const int n = row & (NN - 1);
  float s = ba[t] + pe[n * DD + t];
#pragma unroll
  for (int a = 0; a < 16; ++a) s += xt[(size_t)row * 16 + a] * Wa[a * DD + t];
  X[(size_t)row * DD + t] = s;
}

// edge embed: e = e_t @ Wb + bb + pe[i] + pe[j]
__global__ __launch_bounds__(128) void k_edge_embed(
    const float* __restrict__ et, const float* __restrict__ Wb,
    const float* __restrict__ bb, const float* __restrict__ pe,
    float* __restrict__ E)
{
  const size_t r = blockIdx.x;
  const int t = threadIdx.x;
  const int i = (int)((r >> 7) & (NN - 1)), j = (int)(r & (NN - 1));
  float s = bb[t] + pe[i * DD + t] + pe[j * DD + t];
#pragma unroll
  for (int c = 0; c < 8; ++c) s += et[r * 8 + c] * Wb[c * DD + t];
  E[r * DD + t] = s;
}

// sym_tensor in place: lower triangle (i>j) <- upper value if |upper|>0
__global__ __launch_bounds__(128) void k_sym(float* __restrict__ E)
{
  const size_t r = blockIdx.x;
  const int t = threadIdx.x;
  const int b = (int)(r >> 14), i = (int)((r >> 7) & 127), j = (int)(r & 127);
  if (i <= j) return;
  const size_t up = (((size_t)(b * NN + j)) * NN + i) * DD + t;
  const float u = E[up];
  if (fabsf(u) > 0.0f) E[r * DD + t] = u;
}

// p_t = relu(pet_tab[t]) @ Wp + bp
__global__ __launch_bounds__(128) void k_pet(
    const float* __restrict__ tab, const int* __restrict__ samp,
    const float* __restrict__ Wp, const float* __restrict__ bp,
    float* __restrict__ PT)
{
  const int b = blockIdx.x, t = threadIdx.x;
  const int tt = samp[b];
  float s = bp[t];
  for (int c = 0; c < DD; ++c)
    s += fmaxf(tab[(size_t)tt * DD + c], 0.0f) * Wp[c * DD + t];
  PT[b * DD + t] = s;
}

// e_new += Ni[b,i] + Nj[b,j]   (qkv layout: [b*n][640], Ni@384, Nj@512)
__global__ __launch_bounds__(128) void k_addNiNj(
    float* __restrict__ E, const float* __restrict__ qkv)
{
  const size_t r = blockIdx.x;
  const int t = threadIdx.x;
  const int b = (int)(r >> 14), i = (int)((r >> 7) & 127), j = (int)(r & 127);
  E[r * DD + t] += qkv[(size_t)(b * NN + i) * 640 + 384 + t]
                 + qkv[(size_t)(b * NN + j) * 640 + 512 + t];
}

// att[b,h,i,j] = 0.25 * sum_k Q[bhik] * e_new[bhijk] * K[bhjk]
__global__ __launch_bounds__(128) void k_att(
    const float* __restrict__ qkv, const float* __restrict__ E,
    float* __restrict__ att)
{
  const int blk = blockIdx.x;              // (b*8 + h)*128 + i
  const int t = threadIdx.x;               // j
  const int b = blk >> 10, h = (blk >> 7) & 7, i = blk & 127;
  const size_t qb = (size_t)(b * NN + i) * 640 + h * DHH;
  const size_t kb = (size_t)(b * NN + t) * 640 + 128 + h * DHH;
  const size_t eb = (((size_t)(b * NN + i)) * NN + t) * DD + h * DHH;
  float s = 0.0f;
#pragma unroll
  for (int k = 0; k < DHH; ++k)
    s += qkv[qb + k] * E[eb + k] * qkv[kb + k];
  att[(size_t)blk * NN + t] = s * 0.25f;   // 1/sqrt(16)
}

// softmax over the query index i (axis=2 of [b,h,i,j]); block per (b,h,j)
__global__ __launch_bounds__(128) void k_softmax_q(float* __restrict__ att)
{
  __shared__ float sh[128];
  const int t = threadIdx.x;               // i
  const int bh = blockIdx.x >> 7;
  const int j = blockIdx.x & 127;
  const size_t idx = (size_t)bh * (NN * NN) + (size_t)t * NN + j;
  const float v = att[idx];
  sh[t] = v; __syncthreads();
  for (int s = 64; s > 0; s >>= 1) { if (t < s) sh[t] = fmaxf(sh[t], sh[t + s]); __syncthreads(); }
  const float mx = sh[0]; __syncthreads();
  const float ev = __expf(v - mx);
  sh[t] = ev; __syncthreads();
  for (int s = 64; s > 0; s >>= 1) { if (t < s) sh[t] += sh[t + s]; __syncthreads(); }
  att[idx] = ev / sh[0];
}

// xh[b,i,h*16+k] = sum_j att[b,h,i,j] * V[b,j,h*16+k]
__global__ __launch_bounds__(128) void k_xh(
    const float* __restrict__ att, const float* __restrict__ qkv,
    float* __restrict__ xatt)
{
  const int blk = blockIdx.x;              // b*128 + i
  const int t = threadIdx.x;               // d = h*16+k
  const int b = blk >> 7;
  const int h = t >> 4;
  const size_t ab = ((size_t)(b * HH + h) * NN + (blk & 127)) * NN;
  const size_t vb = (size_t)b * NN * 640 + 256 + t;
  float s = 0.0f;
  for (int j = 0; j < NN; ++j)
    s += att[ab + j] * qkv[vb + (size_t)j * 640];
  xatt[(size_t)blk * DD + t] = s;
}

// decoders
__global__ __launch_bounds__(128) void k_dec_a(
    const float* __restrict__ X, const float* __restrict__ W,
    const float* __restrict__ bias, float* __restrict__ out)
{
  __shared__ float sh[128], ps[128];
  const size_t r = blockIdx.x;
  const int t = threadIdx.x;
  sh[t] = X[r * DD + t]; __syncthreads();
  const int o = t & 15, part = t >> 4;     // 8 parts x 16 d
  float s = 0.0f;
#pragma unroll
  for (int q = 0; q < 16; ++q) { int d = part * 16 + q; s += sh[d] * W[d * 16 + o]; }
  ps[t] = s; __syncthreads();
  if (t < 16) {
    float a = bias[t];
#pragma unroll
    for (int p = 0; p < 8; ++p) a += ps[p * 16 + t];
    out[r * 16 + t] = a;
  }
}

__global__ __launch_bounds__(128) void k_dec_b(
    const float* __restrict__ E, const float* __restrict__ W,
    const float* __restrict__ bias, float* __restrict__ out)
{
  __shared__ float sh[128], ps[128];
  const size_t r = blockIdx.x;
  const int t = threadIdx.x;
  sh[t] = E[r * DD + t]; __syncthreads();
  const int o = t & 7, part = t >> 3;      // 16 parts x 8 d
  float s = 0.0f;
#pragma unroll
  for (int q = 0; q < 8; ++q) { int d = part * 8 + q; s += sh[d] * W[d * 8 + o]; }
  ps[t] = s; __syncthreads();
  if (t < 8) {
    float a = bias[t];
#pragma unroll
    for (int p = 0; p < 16; ++p) a += ps[p * 8 + t];
    out[r * 8 + t] = a;
  }
}

// ---------------------------------------------------------------------------
// Weight prep: pack head-stacked [H,D,DH] -> f16 [D, ld] at column `off`,
// f32->f16 convert, 5-way bias concat.
// ---------------------------------------------------------------------------
__global__ __launch_bounds__(256) void k_pack_headW(
    _Float16* __restrict__ dst, const float* __restrict__ src, int ldN, int off)
{
  const int idx = blockIdx.x * 256 + threadIdx.x;   // < 16384 = H*D*DH
  const int h = idx >> 11, d = (idx >> 4) & 127, k = idx & 15;
  dst[(size_t)d * ldN + off + h * DHH + k] = (_Float16)src[idx];
}

__global__ __launch_bounds__(256) void k_cvt_f16(
    _Float16* __restrict__ dst, const float* __restrict__ src, int n)
{
  const int i = blockIdx.x * 256 + threadIdx.x;
  if (i < n) dst[i] = (_Float16)src[i];
}

__global__ __launch_bounds__(256) void k_concat5(
    float* __restrict__ dst, const float* a, const float* b, const float* c,
    const float* d, const float* e)
{
  const int t = blockIdx.x * 256 + threadIdx.x;
  if (t >= 640) return;
  const float* srcs[5] = {a, b, c, d, e};
  dst[t] = srcs[t >> 7][t & 127];
}

// ---------------------------------------------------------------------------
// Host orchestration
// ---------------------------------------------------------------------------
extern "C" void kernel_launch(void* const* d_in, const int* in_sizes, int n_in,
                              void* d_out, int out_size, void* d_ws, size_t ws_size,
                              hipStream_t stream)
{
  (void)in_sizes; (void)n_in; (void)out_size; (void)ws_size;

  // ---- input leaf map (top-level insertion order; params in pytree order) --
  const float* x_t   = (const float*)d_in[0];
  const float* e_t   = (const float*)d_in[1];
  const int*   samp  = (const int*)  d_in[2];
  const float* atomW = (const float*)d_in[3];
  const float* atomb = (const float*)d_in[4];
  const float* bondW = (const float*)d_in[5];
  const float* bondb = (const float*)d_in[6];
  const float* decaW = (const float*)d_in[7];
  const float* decab = (const float*)d_in[8];
  const float* decbW = (const float*)d_in[9];
  const float* decbb = (const float*)d_in[10];
  const int LBASE = 11;
  const float* pe    = (const float*)d_in[LBASE + 32 * NLAYERS + 0];
  const float* petW  = (const float*)d_in[LBASE + 32 * NLAYERS + 1];
  const float* petb  = (const float*)d_in[LBASE + 32 * NLAYERS + 2];
  const float* ptab  = (const float*)d_in[LBASE + 32 * NLAYERS + 3];

  // ---- workspace carve-up --------------------------------------------------
  char* wp = (char*)d_ws;
  auto take = [&](size_t bytes) -> char* {
    char* p = wp; wp += (bytes + 255) & ~(size_t)255; return p;
  };
  float*    EB1  = (float*)   take((size_t)BNN * DD * 4);     // 134 MB
  float*    EB2  = (float*)   take((size_t)BNN * DD * 4);     // 134 MB
  _Float16* EHID = (_Float16*)take((size_t)CHUNK * FF * 2);   // 32 MB
  float*    ATT  = (float*)   take((size_t)BB * HH * NN * NN * 4);
  float*    QKV  = (float*)   take((size_t)BN * 5 * DD * 4);
  float*    XB1  = (float*)   take((size_t)BN * DD * 4);
  float*    XB2  = (float*)   take((size_t)BN * DD * 4);
  float*    XATT = (float*)   take((size_t)BN * DD * 4);
  _Float16* XHID = (_Float16*)take((size_t)BN * FF * 2);
  float*    PT   = (float*)   take((size_t)BB * DD * 4);
  _Float16* Wqkv = (_Float16*)take((size_t)DD * 5 * DD * 2);
  _Float16* WE   = (_Float16*)take((size_t)DD * DD * 2);
  _Float16* WOx  = (_Float16*)take((size_t)DD * DD * 2);
  _Float16* WOe  = (_Float16*)take((size_t)DD * DD * 2);
  _Float16* Wx1  = (_Float16*)take((size_t)DD * FF * 2);
  _Float16* Wx2  = (_Float16*)take((size_t)FF * DD * 2);
  _Float16* We1  = (_Float16*)take((size_t)DD * FF * 2);
  _Float16* We2  = (_Float16*)take((size_t)FF * DD * 2);
  float*    B640 = (float*)   take(640 * 4);

  // ---- embeddings ----------------------------------------------------------
  k_node_embed<<<BN,  128, 0, stream>>>(x_t, atomW, atomb, pe, XB1);
  k_edge_embed<<<BNN, 128, 0, stream>>>(e_t, bondW, bondb, pe, EB1);
  k_sym       <<<BNN, 128, 0, stream>>>(EB1);
  k_pet       <<<BB,  128, 0, stream>>>(ptab, samp, petW, petb, PT);

  float* S = EB1;   // edge stream at layer entry
  float* T = EB2;

  for (int l = 0; l < NLAYERS; ++l) {
    const float* const* P = (const float* const*)(d_in + LBASE + 32 * l);
    const float *EW = P[0],  *Eb = P[1],  *KW = P[2],  *Kb = P[3];
    const float *NiW = P[4], *Nib = P[5], *NjW = P[6], *Njb = P[7];
    const float *QW = P[8],  *Qb = P[9],  *VW = P[10], *Vb = P[11];
    const float *WOeW = P[12], *WOeb = P[13], *WOxW = P[14], *WOxb = P[15];
    const float *be = P[16], *be2 = P[17], *bx = P[18], *bx2 = P[19];
    const float *ge = P[20], *ge2 = P[21], *gx = P[22], *gx2 = P[23];
    const float *me1W = P[24], *me1b = P[25], *me2W = P[26], *me2b = P[27];
    const float *mx1W = P[28], *mx1b = P[29], *mx2W = P[30], *mx2b = P[31];

    // weight prep (f16 packs)
    k_pack_headW<<<64, 256, 0, stream>>>(Wqkv, QW,  640, 0);
    k_pack_headW<<<64, 256, 0, stream>>>(Wqkv, KW,  640, 128);
    k_pack_headW<<<64, 256, 0, stream>>>(Wqkv, VW,  640, 256);
    k_pack_headW<<<64, 256, 0, stream>>>(Wqkv, NiW, 640, 384);
    k_pack_headW<<<64, 256, 0, stream>>>(Wqkv, NjW, 640, 512);
    k_concat5   <<<3,  256, 0, stream>>>(B640, Qb, Kb, Vb, Nib, Njb);
    k_pack_headW<<<64, 256, 0, stream>>>(WE, EW, 128, 0);
    k_cvt_f16<<<64,  256, 0, stream>>>(WOx, WOxW, DD * DD);
    k_cvt_f16<<<64,  256, 0, stream>>>(WOe, WOeW, DD * DD);
    k_cvt_f16<<<256, 256, 0, stream>>>(Wx1, mx1W, DD * FF);
    k_cvt_f16<<<256, 256, 0, stream>>>(Wx2, mx2W, FF * DD);
    k_cvt_f16<<<256, 256, 0, stream>>>(We1, me1W, DD * FF);
    k_cvt_f16<<<256, 256, 0, stream>>>(We2, me2W, FF * DD);

    // pre-norm (fused + p_t)
    k_layernorm<<<BN,  128, 0, stream>>>(XB1, PT, gx, bx, XB2, NN);        // xn
    k_layernorm<<<BNN, 128, 0, stream>>>(S,   PT, ge, be, T,   NN * NN);   // en

    // node projections Q|K|V|Ni|Nj : [2048,128] @ [128,640]
    k_gemm<float, float, 0, false><<<dim3(BN / 64, 640 / 32), 256, 0, stream>>>(
        XB2, Wqkv, B640, nullptr, QKV, BN, 640, DD);
    // edge E projection : [262144,128] @ [128,128] -> e_new in S (old stream dead)
    k_gemm<float, float, 0, false><<<dim3(BNN / 64, DD / 32), 256, 0, stream>>>(
        T, WE, Eb, nullptr, S, BNN, DD, DD);
    k_addNiNj<<<BNN, 128, 0, stream>>>(S, QKV);

    // attention
    k_att      <<<BB * HH * NN, 128, 0, stream>>>(QKV, S, ATT);
    k_softmax_q<<<BB * HH * NN, 128, 0, stream>>>(ATT);
    k_xh       <<<BN, 128, 0, stream>>>(ATT, QKV, XATT);

    // output projections with residual (residual base = normalized inputs)
    k_gemm<float, float, 0, true><<<dim3(BN / 64, DD / 32), 256, 0, stream>>>(
        XATT, WOx, WOxb, XB2, XB1, BN, DD, DD);                 // x stream
    k_gemm<float, float, 0, true><<<dim3(BNN / 64, DD / 32), 256, 0, stream>>>(
        S, WOe, WOeb, T, T, BNN, DD, DD);                        // e stream -> T

    // MLPs
    k_layernorm<<<BN,  128, 0, stream>>>(XB1, nullptr, gx2, bx2, XB2, NN);
    k_layernorm<<<BNN, 128, 0, stream>>>(T,   nullptr, ge2, be2, S,   NN * NN);
    k_gemm<float, _Float16, 1, false><<<dim3(BN / 64, FF / 32), 256, 0, stream>>>(
        XB2, Wx1, mx1b, nullptr, XHID, BN, FF, DD);
    k_gemm<_Float16, float, 0, true><<<dim3(BN / 64, DD / 32), 256, 0, stream>>>(
        XHID, Wx2, mx2b, XB1, XB1, BN, DD, FF);
    for (int c = 0; c < NCHUNK; ++c) {
      float* Sc = S + (size_t)c * CHUNK * DD;
      float* Tc = T + (size_t)c * CHUNK * DD;
      k_gemm<float, _Float16, 1, false><<<dim3(CHUNK / 64, FF / 32), 256, 0, stream>>>(
          Sc, We1, me1b, nullptr, EHID, CHUNK, FF, DD);
      k_gemm<_Float16, float, 0, true><<<dim3(CHUNK / 64, DD / 32), 256, 0, stream>>>(
          EHID, We2, me2b, Tc, Tc, CHUNK, DD, FF);
    }
    k_sym<<<BNN, 128, 0, stream>>>(T);

    { float* tmp = S; S = T; T = tmp; }   // stream now in (new) S
  }

  // ---- decoders ------------------------------------------------------------
  float* out_a = (float*)d_out;                       // [B,N,16]
  float* out_b = out_a + (size_t)BN * 16;             // [B,N,N,8]
  k_dec_a<<<BN,  128, 0, stream>>>(XB1, decaW, decab, out_a);
  k_dec_b<<<BNN, 128, 0, stream>>>(S,   decbW, decbb, out_b);
}
